// CffmTransformerBlock3d3_78752520339655
// MI455X (gfx1250) — matrix operand order repaired and checked by reading, not compiled, not run on hardware
//
#include <hip/hip_runtime.h>
#include <hip/hip_bf16.h>
#include <stdint.h>

// ---------------------------------------------------------------------------
// CDNA5 (gfx1250) implementation of the CFFM transformer block.
// All matmuls use v_wmma_f32_16x16x32_bf16 (wave32 WMMA, f32 accumulate).
// Flash-style attention (online softmax, DPP16 lane reductions); K/V gathered
// into LDS per (window, head); no score matrix or K_all/V_all in HBM.
// ---------------------------------------------------------------------------

typedef __bf16 v16bf __attribute__((ext_vector_type(16)));
typedef float  v8f   __attribute__((ext_vector_type(8)));

union FragB { v16bf v; unsigned u[8]; };

__device__ __forceinline__ unsigned short f2bf(float f) {
  unsigned u = __float_as_uint(f);
  u += 0x7FFFu + ((u >> 16) & 1u);          // round-to-nearest-even
  return (unsigned short)(u >> 16);
}
__device__ __forceinline__ unsigned pack2bf(float lo, float hi) {
  return (unsigned)f2bf(lo) | ((unsigned)f2bf(hi) << 16);
}
// A-fragment K index for 16-bit 16x32 A tiles (contiguous, matches B):
// lanes 0-15 -> K 0..15 (v0..7); lanes 16-31 -> K 16..31.
__device__ __forceinline__ int akk(int i, int sel) {
  return 16 * sel + 2 * i;
}

// DPP16 cross-lane ops: xor1 / xor2 / xor7(row_half_mirror) / xor15(row_mirror)
// give a full 16-lane reduction tree with pure VALU (no ds_bpermute).
template <int CTRL>
__device__ __forceinline__ float dppf(float v) {
  return __int_as_float(__builtin_amdgcn_update_dpp(
      0, __float_as_int(v), CTRL, 0xF, 0xF, true));
}
__device__ __forceinline__ float rowmax16(float v) {
  v = fmaxf(v, dppf<0xB1>(v));   // quad_perm [1,0,3,2]  (xor 1)
  v = fmaxf(v, dppf<0x4E>(v));   // quad_perm [2,3,0,1]  (xor 2)
  v = fmaxf(v, dppf<0x141>(v));  // row_half_mirror      (xor 7 -> other quad)
  v = fmaxf(v, dppf<0x140>(v));  // row_mirror           (xor 15 -> other 8)
  return v;
}
__device__ __forceinline__ float rowsum16(float v) {
  v += dppf<0xB1>(v);
  v += dppf<0x4E>(v);
  v += dppf<0x141>(v);
  v += dppf<0x140>(v);
  return v;
}

// ---------------------------------------------------------------------------
// Generic GEMM: C[M,N] = A[M,256] @ W[:, wcol0 : wcol0+N] + bias
// A is f32 or bf16 (template), W/bias f32, C is bf16 or f32 (template).
// Block = 128 threads (4 waves), tile 64x64, K-step 32 (8 iterations).
// Vectorized staging: float4/uint2 global loads, packed b32/b64 LDS stores.
// ---------------------------------------------------------------------------
template <bool A_BF16, bool OUT_BF16>
__global__ __launch_bounds__(128) void gemm_wmma(
    const void* __restrict__ Ap, const float* __restrict__ W,
    const float* __restrict__ bias, void* __restrict__ Cp, int N, int ldw,
    int wcol0) {
  __shared__ __align__(16) unsigned short As[64][32];  // [m][k]
  __shared__ __align__(16) unsigned short Bs[64][32];  // [n][k] pairs in k
  const int tid = threadIdx.x;
  const int wave = tid >> 5, lane = tid & 31;
  const int sel = lane >> 4, lm = lane & 15;
  const int m0 = blockIdx.x * 64, n0 = blockIdx.y * 64;

  v8f acc[4];
#pragma unroll
  for (int t = 0; t < 4; ++t)
#pragma unroll
    for (int r = 0; r < 8; ++r) acc[t][r] = 0.f;

  for (int kt = 0; kt < 8; ++kt) {
    // stage A tile (64x32): 4 x 16B loads per thread, packed b64 LDS stores
#pragma unroll
    for (int e = 0; e < 4; ++e) {
      const int u = e * 128 + tid;  // 0..511 units of 4 elements
      const int m = u >> 3, k4 = (u & 7) * 4;
      uint2 p;
      if (A_BF16) {
        p = *(const uint2*)&(
            (const unsigned short*)Ap)[(size_t)(m0 + m) * 256 + kt * 32 + k4];
      } else {
        const float4 f = *(const float4*)&(
            (const float*)Ap)[(size_t)(m0 + m) * 256 + kt * 32 + k4];
        p.x = pack2bf(f.x, f.y);
        p.y = pack2bf(f.z, f.w);
      }
      *(uint2*)&As[m][k4] = p;
    }
    // stage W tile (32x64) transposed: unit = (2 k-rows) x (4 n), b32 stores
#pragma unroll
    for (int e = 0; e < 2; ++e) {
      const int u = e * 128 + tid;  // 0..255
      const int kp = u >> 4, n4 = (u & 15) * 4;
      const float* wp = &W[(size_t)(kt * 32 + 2 * kp) * ldw + wcol0 + n0 + n4];
      const float4 f0 = *(const float4*)wp;
      const float4 f1 = *(const float4*)(wp + ldw);
      *(unsigned*)&Bs[n4 + 0][2 * kp] = pack2bf(f0.x, f1.x);
      *(unsigned*)&Bs[n4 + 1][2 * kp] = pack2bf(f0.y, f1.y);
      *(unsigned*)&Bs[n4 + 2][2 * kp] = pack2bf(f0.z, f1.z);
      *(unsigned*)&Bs[n4 + 3][2 * kp] = pack2bf(f0.w, f1.w);
    }
    __syncthreads();

    FragB af;
    const int m = wave * 16 + lm;
#pragma unroll
    for (int i = 0; i < 8; ++i)
      af.u[i] = *(const unsigned*)&As[m][akk(i, sel)];
#pragma unroll
    for (int t = 0; t < 4; ++t) {
      FragB bf;
      const int n = t * 16 + lm;
#pragma unroll
      for (int j = 0; j < 8; ++j)
        bf.u[j] = *(const unsigned*)&Bs[n][16 * sel + 2 * j];
      acc[t] = __builtin_amdgcn_wmma_f32_16x16x32_bf16(
          false, af.v, false, bf.v, (short)0, acc[t], false, false);
    }
    __syncthreads();
  }

  // epilogue: bias + store
#pragma unroll
  for (int t = 0; t < 4; ++t) {
    const int col = n0 + t * 16 + lm;
    const float bv = bias[wcol0 + col];
#pragma unroll
    for (int r = 0; r < 8; ++r) {
      const int row = m0 + wave * 16 + r + 8 * sel;
      const float v = acc[t][r] + bv;
      if (OUT_BF16)
        ((unsigned short*)Cp)[(size_t)row * N + col] = f2bf(v);
      else
        ((float*)Cp)[(size_t)row * N + col] = v;
    }
  }
}

// ---------------------------------------------------------------------------
// Attention: one block per (window, head). 128 threads = 4 waves, each wave
// owns 16 query rows (49 real queries padded to 64). 377 keys padded to 384.
// ---------------------------------------------------------------------------
#define LPAD 384
#define LKEYS 377

__global__ __launch_bounds__(128) void attn_wmma(
    const float* __restrict__ pool0, const unsigned short* __restrict__ QKVx,
    const unsigned short* __restrict__ KV1,
    const unsigned short* __restrict__ KV2,
    const unsigned short* __restrict__ KV3, unsigned short* __restrict__ Obuf) {
  __shared__ __align__(16) unsigned short Qs[64][32];     // Q (unscaled) bf16
  __shared__ __align__(16) unsigned short Ks[LPAD][32];   // K  [key][d]
  __shared__ __align__(16) unsigned short Vt[32][LPAD];   // V^T [d][key]
  __shared__ __align__(16) unsigned short Ps[4][16][32];  // per-wave P slab

  const int tid = threadIdx.x;
  const int wave = tid >> 5, lane = tid & 31;
  const int sel = lane >> 4, lm = lane & 15;
  const int w = blockIdx.x >> 3, head = blockIdx.x & 7;
  const int b = w >> 8, wi = (w >> 4) & 15, wj = w & 15;
  const float scale = 0.17677669529663687f;  // 32^-0.5 (folded into S)

  // ---- stage Q: 2 threads per row, pure 16B copies ----
  {
    const int row = tid >> 1, half = (tid & 1) * 16;
    if (row < 49) {
      const size_t pix =
          ((size_t)(b * 112 + wi * 7 + row / 7)) * 112 + (wj * 7 + row % 7);
      const unsigned short* src = &QKVx[pix * 768 + head * 32 + half];
      *(uint4*)&Qs[row][half] = *(const uint4*)&src[0];
      *(uint4*)&Qs[row][half + 8] = *(const uint4*)&src[8];
    } else {
      const uint4 z = {0u, 0u, 0u, 0u};
      *(uint4*)&Qs[row][half] = z;
      *(uint4*)&Qs[row][half + 8] = z;
    }
  }

  // ---- gather K / V: 4 threads per key, 8 dims each ----
  {
    const int kq = tid >> 2;        // key slot within pass (0..31)
    const int dq = (tid & 3) * 8;   // dim offset 0/8/16/24
    for (int pass = 0; pass < 12; ++pass) {
      const int j = pass * 32 + kq;  // 0..383
      const unsigned short* kp = nullptr;
      const unsigned short* vp = nullptr;
      const float* fp = nullptr;
      if (j < 49) {  // self window
        const size_t pix =
            ((size_t)(b * 112 + wi * 7 + j / 7)) * 112 + (wj * 7 + j % 7);
        kp = &QKVx[pix * 768 + 256 + head * 32];
        vp = kp + 256;
      } else if (j < 245) {  // 4 rolled shifts of +-EXPAND
        const int jj = j - 49, s = jj / 49, p = jj % 49;
        const int sh = (s >> 1) ? 3 : -3;
        const int sw = (s & 1) ? 3 : -3;
        const int hh = (wi * 7 + p / 7 - sh + 112) % 112;
        const int cc = (wj * 7 + p % 7 - sw + 112) % 112;
        const size_t pix = ((size_t)(b * 112 + hh)) * 112 + cc;
        kp = &QKVx[pix * 768 + 256 + head * 32];
        vp = kp + 256;
      } else if (j < 294) {  // pool0 raw (flat reinterpret to (512,8,49,32))
        fp = &pool0[(size_t)w * 12544 + head * 1568 + (size_t)(j - 245) * 32];
      } else if (j < 343) {  // clip1 K/V
        const size_t row = (size_t)w * 49 + (j - 294);
        kp = &KV1[row * 512 + head * 32];
        vp = kp + 256;
      } else if (j < 368) {  // clip2 K/V
        const size_t row = (size_t)w * 25 + (j - 343);
        kp = &KV2[row * 512 + head * 32];
        vp = kp + 256;
      } else if (j < LKEYS) {  // clip3 K/V
        const size_t row = (size_t)w * 9 + (j - 368);
        kp = &KV3[row * 512 + head * 32];
        vp = kp + 256;
      }
      uint4 kword = {0u, 0u, 0u, 0u};
      uint4 vword = {0u, 0u, 0u, 0u};
      if (fp) {
        const float4 f0 = *(const float4*)&fp[dq];
        const float4 f1 = *(const float4*)&fp[dq + 4];
        kword.x = pack2bf(f0.x, f0.y);
        kword.y = pack2bf(f0.z, f0.w);
        kword.z = pack2bf(f1.x, f1.y);
        kword.w = pack2bf(f1.z, f1.w);
        vword = kword;
      } else if (kp) {
        kword = *(const uint4*)&kp[dq];
        vword = *(const uint4*)&vp[dq];
      }
      *(uint4*)&Ks[j][dq] = kword;  // one b128 store
      const unsigned vv[4] = {vword.x, vword.y, vword.z, vword.w};
#pragma unroll
      for (int i = 0; i < 4; ++i) {
        Vt[dq + 2 * i + 0][j] = (unsigned short)(vv[i] & 0xFFFFu);
        Vt[dq + 2 * i + 1][j] = (unsigned short)(vv[i] >> 16);
      }
    }
  }
  __syncthreads();

  // Q fragment for this wave's 16 query rows
  FragB qf;
  {
    const int m = wave * 16 + lm;
#pragma unroll
    for (int i = 0; i < 8; ++i)
      qf.u[i] = *(const unsigned*)&Qs[m][akk(i, sel)];
  }

  float rmax[8], rsum[8];
#pragma unroll
  for (int r = 0; r < 8; ++r) {
    rmax[r] = -1e30f;
    rsum[r] = 0.f;
  }
  v8f o0, o1;
#pragma unroll
  for (int r = 0; r < 8; ++r) {
    o0[r] = 0.f;
    o1[r] = 0.f;
  }

  for (int c = 0; c < 12; ++c) {  // 12 chunks of 32 keys
    v8f s0, s1;
#pragma unroll
    for (int r = 0; r < 8; ++r) {
      s0[r] = 0.f;
      s1[r] = 0.f;
    }
    // S = Q @ K^T  (B fragment: [d][key], d pairs contiguous in Ks rows)
    {
      FragB kf;
      const int key = c * 32 + lm;
#pragma unroll
      for (int jv = 0; jv < 8; ++jv)
        kf.u[jv] = *(const unsigned*)&Ks[key][16 * sel + 2 * jv];
      s0 = __builtin_amdgcn_wmma_f32_16x16x32_bf16(false, qf.v, false, kf.v,
                                                   (short)0, s0, false, false);
    }
    {
      FragB kf;
      const int key = c * 32 + 16 + lm;
#pragma unroll
      for (int jv = 0; jv < 8; ++jv)
        kf.u[jv] = *(const unsigned*)&Ks[key][16 * sel + 2 * jv];
      s1 = __builtin_amdgcn_wmma_f32_16x16x32_bf16(false, qf.v, false, kf.v,
                                                   (short)0, s1, false, false);
    }
    // scale + mask padding keys
#pragma unroll
    for (int r = 0; r < 8; ++r) {
      s0[r] *= scale;
      s1[r] *= scale;
    }
    if (c * 32 + lm >= LKEYS)
#pragma unroll
      for (int r = 0; r < 8; ++r) s0[r] = -1e30f;
    if (c * 32 + 16 + lm >= LKEYS)
#pragma unroll
      for (int r = 0; r < 8; ++r) s1[r] = -1e30f;

    // online softmax; DPP16 reductions over the 16 lanes of each half-wave
    float fac[8];
#pragma unroll
    for (int r = 0; r < 8; ++r) {
      const float m2 = rowmax16(fmaxf(s0[r], s1[r]));
      const float nm = fmaxf(rmax[r], m2);
      fac[r] = __expf(rmax[r] - nm);
      rmax[r] = nm;
      s0[r] = __expf(s0[r] - nm);
      s1[r] = __expf(s1[r] - nm);
      const float cs = rowsum16(s0[r] + s1[r]);
      rsum[r] = rsum[r] * fac[r] + cs;
      o0[r] *= fac[r];
      o1[r] *= fac[r];
    }

    // P: D-layout -> A-layout via the per-wave LDS slab. DS ops from one
    // wave complete in order, so no block barrier is needed -- only a
    // scheduling fence around the slab reuse.
    __builtin_amdgcn_wave_barrier();
#pragma unroll
    for (int r = 0; r < 8; ++r) {
      Ps[wave][r + 8 * sel][lm] = f2bf(s0[r]);
      Ps[wave][r + 8 * sel][16 + lm] = f2bf(s1[r]);
    }
    __builtin_amdgcn_wave_barrier();
    FragB pf;
#pragma unroll
    for (int i = 0; i < 8; ++i)
      pf.u[i] = *(const unsigned*)&Ps[wave][lm][akk(i, sel)];

    // O += P @ V  (B fragment from Vt: key pairs contiguous in Vt rows)
    {
      FragB vf;
#pragma unroll
      for (int jv = 0; jv < 8; ++jv)
        vf.u[jv] = *(const unsigned*)&Vt[lm][c * 32 + 16 * sel + 2 * jv];
      o0 = __builtin_amdgcn_wmma_f32_16x16x32_bf16(false, pf.v, false, vf.v,
                                                   (short)0, o0, false, false);
    }
    {
      FragB vf;
#pragma unroll
      for (int jv = 0; jv < 8; ++jv)
        vf.u[jv] = *(const unsigned*)&Vt[16 + lm][c * 32 + 16 * sel + 2 * jv];
      o1 = __builtin_amdgcn_wmma_f32_16x16x32_bf16(false, pf.v, false, vf.v,
                                                   (short)0, o1, false, false);
    }
  }

  // normalize and store (row = w*49 + q, col = head*32 + d) as bf16
#pragma unroll
  for (int r = 0; r < 8; ++r) {
    const int m = wave * 16 + r + 8 * sel;
    if (m < 49) {
      const float inv = 1.0f / rsum[r];
      const size_t row = (size_t)w * 49 + m;
      Obuf[row * 256 + head * 32 + lm] = f2bf(o0[r] * inv);
      Obuf[row * 256 + head * 32 + 16 + lm] = f2bf(o1[r] * inv);
    }
  }
}

// ---------------------------------------------------------------------------
// Host-side orchestration.
// Workspace layout (bytes):
//   QKVx : 25088*768*2 = 38,535,168
//   KV1  : 25088*512*2 = 25,690,112
//   KV2  : 12800*512*2 = 13,107,200
//   KV3  :  4608*512*2 =  4,718,592
//   Obuf : 25088*256*2 = 12,845,056   (total ~94.9 MB)
// ---------------------------------------------------------------------------
extern "C" void kernel_launch(void* const* d_in, const int* in_sizes, int n_in,
                              void* d_out, int out_size, void* d_ws,
                              size_t ws_size, hipStream_t stream) {
  const float* x = (const float*)d_in[0];
  const float* pool0 = (const float*)d_in[1];
  const float* clip1 = (const float*)d_in[2];
  const float* clip2 = (const float*)d_in[3];
  const float* clip3 = (const float*)d_in[4];
  const float* qkv_w = (const float*)d_in[5];
  const float* qkv_b = (const float*)d_in[6];
  const float* proj_w = (const float*)d_in[7];
  const float* proj_b = (const float*)d_in[8];

  uint8_t* ws = (uint8_t*)d_ws;
  unsigned short* QKVx = (unsigned short*)(ws);
  unsigned short* KV1 = (unsigned short*)(ws + 38535168u);
  unsigned short* KV2 = (unsigned short*)(ws + 64225280u);
  unsigned short* KV3 = (unsigned short*)(ws + 77332480u);
  unsigned short* Obuf = (unsigned short*)(ws + 82051072u);

  const dim3 blk(128);
  // QKV for x: (25088,256) @ (256,768)
  gemm_wmma<false, true><<<dim3(392, 12), blk, 0, stream>>>(
      x, qkv_w, qkv_b, QKVx, 768, 768, 0);
  // K/V only for clips: (T,256) @ qkv_w[:,256:768]
  gemm_wmma<false, true><<<dim3(392, 8), blk, 0, stream>>>(
      clip1, qkv_w, qkv_b, KV1, 512, 768, 256);
  gemm_wmma<false, true><<<dim3(200, 8), blk, 0, stream>>>(
      clip2, qkv_w, qkv_b, KV2, 512, 768, 256);
  gemm_wmma<false, true><<<dim3(72, 8), blk, 0, stream>>>(
      clip3, qkv_w, qkv_b, KV3, 512, 768, 256);
  // flash attention over 512 windows x 8 heads
  attn_wmma<<<dim3(4096), blk, 0, stream>>>(pool0, QKVx, KV1, KV2, KV3, Obuf);
  // output projection: (25088,256) @ (256,256) -> f32 d_out
  gemm_wmma<true, false><<<dim3(392, 4), blk, 0, stream>>>(
      Obuf, proj_w, proj_b, d_out, 256, 256, 0);
}
